// VQVAE_89395449299400
// MI455X (gfx1250) — matrix operand order repaired and checked by reading, not compiled, not run on hardware
//
#include <hip/hip_runtime.h>

// ---------------------------------------------------------------------------
// Problem constants (from reference): B=16384, F=4096, H=512, D=64, K=512, M=8
// ---------------------------------------------------------------------------
#define BB 16384
#define FF 4096
#define HH 512
#define DD 64
#define KK 512
#define MM 8

typedef __attribute__((ext_vector_type(16))) __bf16 v16bf;
typedef __attribute__((ext_vector_type(2)))  __bf16 v2bf;
typedef __attribute__((ext_vector_type(8)))  float  v8f;
typedef __attribute__((ext_vector_type(4)))  int    v4i;

// gfx1250 async global->LDS copy path (ASYNCcnt-tracked), guarded so the file
// still compiles if the toolchain lacks the builtins.
#if defined(__gfx1250__) && __has_builtin(__builtin_amdgcn_global_load_async_to_lds_b128) && \
    __has_builtin(__builtin_amdgcn_s_wait_asynccnt)
#define HAVE_ASYNC 1
#else
#define HAVE_ASYNC 0
#endif

#if HAVE_ASYNC
// Builtin signature (from clang diagnostic): param0 = v4i* in global AS(1).
typedef __attribute__((address_space(1))) v4i* g_v4i_p;
typedef __attribute__((address_space(3))) v4i* l_v4i_p;
__device__ __forceinline__ void async_g2l_b128(const void* g, void* l) {
    __builtin_amdgcn_global_load_async_to_lds_b128(
        (g_v4i_p)(unsigned long long)g,
        (l_v4i_p)(unsigned long long)l,   // low 32 bits of generic LDS ptr = LDS offset
        0, 0);
}
#endif

// Hardware fp32 -> bf16 (RTNE via v_cvt_bf16_f32 / v_cvt_pk_bf16_f32)
__device__ __forceinline__ unsigned short f2bf(float x) {
    union { __bf16 b; unsigned short s; } u;
    u.b = (__bf16)x;
    return u.s;
}
// Packed pair: maps directly onto v_cvt_pk_bf16_f32
__device__ __forceinline__ unsigned int f2bf2(float x, float y) {
    union { v2bf v; unsigned int u; } p;
    p.v = (v2bf){(__bf16)x, (__bf16)y};
    return p.u;
}

// ---------------------------------------------------------------------------
// Weight convert: fp32 row-major (Kd x N)  ->  bf16 transposed (N x Kd)
// ---------------------------------------------------------------------------
__global__ void vqvae_cvt_wT_bf16(const float* __restrict__ W,
                                  unsigned short* __restrict__ Wt,
                                  int Kd, int N) {
    int i = blockIdx.x * blockDim.x + threadIdx.x;
    if (i >= Kd * N) return;
    int n = i / Kd;
    int k = i - n * Kd;
    Wt[i] = f2bf(W[(size_t)k * N + n]);
}

// ---------------------------------------------------------------------------
// Codebook column squared norms: w2[k] = sum_d cb[d][k]^2
// ---------------------------------------------------------------------------
__global__ void vqvae_w2_kernel(const float* __restrict__ cb, float* __restrict__ w2) {
    int k = blockIdx.x * blockDim.x + threadIdx.x;
    if (k >= KK) return;
    float s = 0.f;
    for (int d = 0; d < DD; ++d) {
        float c = cb[(size_t)d * KK + k];
        s = fmaf(c, c, s);
    }
    w2[k] = s;
}

// ---------------------------------------------------------------------------
// Tiled bf16 WMMA GEMM:  out = act(A @ W + bias)
//   A  : Mrows x Kd   fp32 (A_BF16=0) or bf16 (A_BF16=1)
//   Wt : N x Kd       bf16 (pre-transposed weights)
//   out: Mrows x N    fp32 (C_BF16=0, ptr Cf) or bf16 (C_BF16=1, ptr Cb)
// Tile: 128x64 per 256-thread workgroup; 8 waves in 4(M) x 2(N) grid;
// each wave: 32x32 via 2x2 v_wmma_f32_16x16x32_bf16 accumulators; TK=32.
// bf16 operands are staged into LDS with GLOBAL_LOAD_ASYNC_TO_LDS_B128.
// ACT: 0 = leaky-relu(0.01), 1 = sigmoid
// ---------------------------------------------------------------------------
#define TM 128
#define TN 64
#define TK 32
#define LDA 40   // 80B rows: 16B-aligned chunks, conflict-free (20*row mod 64)
#define LDB 40   // 80B rows: 16B-aligned chunks, conflict-free (20*col mod 64)

template <int ACT, int A_BF16, int C_BF16>
__global__ __launch_bounds__(256) void vqvae_gemm_bias_act(
    const float* __restrict__ Af,
    const unsigned short* __restrict__ Ab,
    const unsigned short* __restrict__ Wt,
    const float* __restrict__ bias,
    float* __restrict__ Cf,
    unsigned short* __restrict__ Cb,
    int Mrows, int Kd, int N)
{
    __shared__ unsigned short As[TM * LDA];
    __shared__ unsigned short Bs[TN * LDB];

    const int tid   = threadIdx.x;
    const int wave  = tid >> 5;
    const int lane  = tid & 31;
    const int half  = lane >> 4;
    const int l16   = lane & 15;
    const int warpM = wave & 3;   // 0..3
    const int warpN = wave >> 2;  // 0..1

    const int m0 = blockIdx.y * TM;
    const int n0 = blockIdx.x * TN;

    v8f acc[2][2];
    for (int i = 0; i < 2; ++i)
        for (int j = 0; j < 2; ++j)
            acc[i][j] = (v8f){0.f, 0.f, 0.f, 0.f, 0.f, 0.f, 0.f, 0.f};

    union Frag { v16bf v; unsigned long long u[4]; uint4 q[2]; };

    for (int k0 = 0; k0 < Kd; k0 += TK) {
        // ---- stage A tile (128 rows x 32 bf16) into LDS ----
        if (A_BF16) {
#if HAVE_ASYNC
            // 512 x 16B chunks, 2 per thread, direct memory->LDS (ASYNCcnt)
            #pragma unroll
            for (int it = 0; it < 2; ++it) {
                int c   = tid + 256 * it;
                int row = c >> 2;
                int q8  = (c & 3) << 3;        // 8 bf16 = 16B
                async_g2l_b128(Ab + (size_t)(m0 + row) * Kd + k0 + q8,
                               &As[row * LDA + q8]);
            }
#else
            int row = tid >> 1;
            int seg = tid & 1;
            const unsigned long long* src =
                (const unsigned long long*)(Ab + (size_t)(m0 + row) * Kd + k0 + seg * 16);
            unsigned long long v0 = src[0];
            unsigned long long v1 = src[1];
            unsigned long long v2 = src[2];
            unsigned long long v3 = src[3];
            unsigned long long* dst = (unsigned long long*)&As[row * LDA + seg * 16];
            dst[0] = v0; dst[1] = v1; dst[2] = v2; dst[3] = v3;
#endif
        } else {
            // fp32 source: hw-convert to bf16 while staging (layer 1 only)
            #pragma unroll
            for (int it = 0; it < 4; ++it) {
                int v   = tid + 256 * it;       // float4 index, 1024 total
                int row = v >> 3;
                int kc  = (v & 7) << 2;
                float4 f = *(const float4*)(Af + (size_t)(m0 + row) * Kd + k0 + kc);
                unsigned int lo = f2bf2(f.x, f.y);   // v_cvt_pk_bf16_f32
                unsigned int hi = f2bf2(f.z, f.w);
                unsigned long long pk = (unsigned long long)lo
                                      | ((unsigned long long)hi << 32);
                *(unsigned long long*)&As[row * LDA + kc] = pk;
            }
        }
        // ---- stage B^T tile (64 rows x 32 bf16) into LDS ----
        {
            int row = tid >> 2;                 // 0..63
            int q8  = (tid & 3) << 3;           // 8 bf16 = 16B
#if HAVE_ASYNC
            async_g2l_b128(Wt + (size_t)(n0 + row) * Kd + k0 + q8,
                           &Bs[row * LDB + q8]);
#else
            const unsigned long long* src =
                (const unsigned long long*)(Wt + (size_t)(n0 + row) * Kd + k0 + q8);
            unsigned long long v0 = src[0];
            unsigned long long v1 = src[1];
            *(unsigned long long*)&Bs[row * LDB + q8]     = v0;
            *(unsigned long long*)&Bs[row * LDB + q8 + 4] = v1;
#endif
        }
#if !HAVE_ASYNC
        // prefetch next K tile while we compute this one
        if (k0 + TK < Kd) {
            if (A_BF16)
                __builtin_prefetch(Ab + (size_t)(m0 + (tid >> 1)) * Kd + k0 + TK, 0, 1);
            else
                __builtin_prefetch(Af + (size_t)(m0 + (tid >> 1)) * Kd + k0 + TK, 0, 1);
            if (tid < TN)
                __builtin_prefetch(Wt + (size_t)(n0 + tid) * Kd + k0 + TK, 0, 1);
        }
#endif
#if HAVE_ASYNC
        __builtin_amdgcn_s_wait_asynccnt(0);   // this wave's async copies landed
#endif
        __syncthreads();                        // whole WG's tiles visible

        // ---- load fragments (CDNA5 16-bit A 16x32 / B 32x16 layouts) ----
        Frag af[2], bfm[2];
        #pragma unroll
        for (int i = 0; i < 2; ++i) {
            int row = warpM * 32 + i * 16 + l16;
            const unsigned short* pa = &As[row * LDA + half * 8];
            af[i].u[0] = *(const unsigned long long*)(pa);
            af[i].u[1] = *(const unsigned long long*)(pa + 4);
            af[i].u[2] = *(const unsigned long long*)(pa + 16);
            af[i].u[3] = *(const unsigned long long*)(pa + 20);
        }
        #pragma unroll
        for (int j = 0; j < 2; ++j) {
            int col = warpN * 32 + j * 16 + l16;
            const unsigned short* pb = &Bs[col * LDB + half * 8];
            bfm[j].q[0] = *(const uint4*)(pb);       // ds_load_b128
            bfm[j].q[1] = *(const uint4*)(pb + 16);  // ds_load_b128
        }
        #pragma unroll
        for (int i = 0; i < 2; ++i)
            #pragma unroll
            for (int j = 0; j < 2; ++j)
                acc[i][j] = __builtin_amdgcn_wmma_f32_16x16x32_bf16(
                    false, af[i].v, false, bfm[j].v,
                    (short)0, acc[i][j], false, false);
        __syncthreads();
    }

    // ---- epilogue: bias + activation, fp32 or bf16 store ----
    #pragma unroll
    for (int i = 0; i < 2; ++i) {
        #pragma unroll
        for (int j = 0; j < 2; ++j) {
            int col = n0 + warpN * 32 + j * 16 + l16;
            float bv = bias[col];
            #pragma unroll
            for (int v = 0; v < 8; ++v) {
                int row = m0 + warpM * 32 + i * 16 + v + 8 * half;
                float r = acc[i][j][v] + bv;
                if (ACT == 0) {
                    r = (r > 0.f) ? r : 0.01f * r;
                } else {
                    r = 1.0f / (1.0f + __expf(-r));
                }
                if (C_BF16) {
                    Cb[(size_t)row * N + col] = f2bf(r);
                } else {
                    Cf[(size_t)row * N + col] = r;
                }
            }
        }
    }
}

// ---------------------------------------------------------------------------
// VQ nearest-codebook: one block per sample b.
// z_e row (512 fp32) cached in LDS; wave m (of 8) scans 512 codes;
// wave32 shfl_xor argmin with lowest-index tiebreak; winners scatter the
// codebook column into z_q (fp32 out), emb (fp32 out) and zq_bf (bf16 for
// the decoder GEMM).
// ---------------------------------------------------------------------------
__global__ __launch_bounds__(256) void vqvae_vq_kernel(
    const float* __restrict__ ze,    // B x 512
    const float* __restrict__ cb,    // 64 x 512
    const float* __restrict__ w2,    // 512
    float* __restrict__ idx_out,     // B x 8 (stored as float)
    float* __restrict__ zq_out,      // B x 512
    float* __restrict__ emb_out,     // B x 512
    unsigned short* __restrict__ zq_bf) // B x 512 bf16
{
    __shared__ float zs[HH];
    __shared__ int   kbest_s[MM];

    const int b   = blockIdx.x;
    const int tid = threadIdx.x;

    zs[tid]       = ze[(size_t)b * HH + tid];
    zs[tid + 256] = ze[(size_t)b * HH + tid + 256];
    __syncthreads();

    const int m    = tid >> 5;   // code position 0..7 (one wave per m)
    const int lane = tid & 31;

    float x2 = 0.f;
    for (int d = 0; d < DD; ++d) {
        float z = zs[d * MM + m];        // LDS broadcast within the wave
        x2 = fmaf(z, z, x2);
    }

    float bestd = 3.402823e38f;
    int   bestk = 0;
    for (int kk = 0; kk < KK / 32; ++kk) {
        int k = lane + (kk << 5);        // coalesced codebook columns
        float dot = 0.f;
        for (int d = 0; d < DD; ++d)
            dot = fmaf(zs[d * MM + m], cb[(size_t)d * KK + k], dot);
        float dist = x2 - 2.0f * dot + w2[k];
        if (dist < bestd) { bestd = dist; bestk = k; }
    }
    // wave32 argmin reduction (lower index wins ties, matching argmin)
    for (int off = 16; off > 0; off >>= 1) {
        float od = __shfl_xor(bestd, off, 32);
        int   ok = __shfl_xor(bestk, off, 32);
        if (od < bestd || (od == bestd && ok < bestk)) { bestd = od; bestk = ok; }
    }
    if (lane == 0) {
        idx_out[(size_t)b * MM + m] = (float)bestk;
        kbest_s[m] = bestk;
    }
    __syncthreads();

    // scatter quantized vector: thread covers (d, m) pairs
    const int mm = tid & 7;
    const int d0 = tid >> 3;             // 0..31
    const int kq = kbest_s[mm];
    {
        float q  = cb[(size_t)d0 * KK + kq];
        size_t o = (size_t)b * HH + d0 * MM + mm;
        zq_out[o]  = q;
        emb_out[o] = q;
        zq_bf[o]   = f2bf(q);
    }
    {
        int d1   = d0 + 32;
        float q  = cb[(size_t)d1 * KK + kq];
        size_t o = (size_t)b * HH + d1 * MM + mm;
        zq_out[o]  = q;
        emb_out[o] = q;
        zq_bf[o]   = f2bf(q);
    }
}

// ---------------------------------------------------------------------------
// Host launcher
// ---------------------------------------------------------------------------
extern "C" void kernel_launch(void* const* d_in, const int* in_sizes, int n_in,
                              void* d_out, int out_size, void* d_ws, size_t ws_size,
                              hipStream_t stream) {
    const float* x   = (const float*)d_in[0];
    const float* We1 = (const float*)d_in[1];
    const float* be1 = (const float*)d_in[2];
    const float* We2 = (const float*)d_in[3];
    const float* be2 = (const float*)d_in[4];
    const float* We3 = (const float*)d_in[5];
    const float* be3 = (const float*)d_in[6];
    const float* Wd1 = (const float*)d_in[7];
    const float* bd1 = (const float*)d_in[8];
    const float* Wd2 = (const float*)d_in[9];
    const float* bd2 = (const float*)d_in[10];
    const float* Wd3 = (const float*)d_in[11];
    const float* bd3 = (const float*)d_in[12];
    const float* cb  = (const float*)d_in[13];

    // output layout (flat, return order): idx, z_e, z_q, emb, X_pred
    float* out      = (float*)d_out;
    float* idx_out  = out;                               // 131072
    float* ze_out   = out + (size_t)BB * MM;             // 8388608
    float* zq_out   = ze_out + (size_t)BB * HH;
    float* emb_out  = zq_out + (size_t)BB * HH;
    float* xp_out   = emb_out + (size_t)BB * HH;

    // workspace carve-up (bf16 weights transposed + bf16 activations)
    char* p = (char*)d_ws;
    unsigned short* We1t = (unsigned short*)p; p += (size_t)FF * 256 * 2;
    unsigned short* We2t = (unsigned short*)p; p += (size_t)256 * 128 * 2;
    unsigned short* We3t = (unsigned short*)p; p += (size_t)128 * HH * 2;
    unsigned short* Wd1t = (unsigned short*)p; p += (size_t)HH * 128 * 2;
    unsigned short* Wd2t = (unsigned short*)p; p += (size_t)128 * 256 * 2;
    unsigned short* Wd3t = (unsigned short*)p; p += (size_t)256 * FF * 2;
    float* w2v = (float*)p;            p += (size_t)KK * 4;
    unsigned short* h1b  = (unsigned short*)p; p += (size_t)BB * 256 * 2;
    unsigned short* h2b  = (unsigned short*)p; p += (size_t)BB * 128 * 2;
    unsigned short* zqb  = (unsigned short*)p; p += (size_t)BB * HH * 2;
    unsigned short* g1b  = (unsigned short*)p; p += (size_t)BB * 128 * 2;
    unsigned short* g2b  = (unsigned short*)p; p += (size_t)BB * 256 * 2;

    // 1) weight conversions (fp32 KxN -> bf16 NxK)
    vqvae_cvt_wT_bf16<<<(FF * 256 + 255) / 256, 256, 0, stream>>>(We1, We1t, FF, 256);
    vqvae_cvt_wT_bf16<<<(256 * 128 + 255) / 256, 256, 0, stream>>>(We2, We2t, 256, 128);
    vqvae_cvt_wT_bf16<<<(128 * HH + 255) / 256, 256, 0, stream>>>(We3, We3t, 128, HH);
    vqvae_cvt_wT_bf16<<<(HH * 128 + 255) / 256, 256, 0, stream>>>(Wd1, Wd1t, HH, 128);
    vqvae_cvt_wT_bf16<<<(128 * 256 + 255) / 256, 256, 0, stream>>>(Wd2, Wd2t, 128, 256);
    vqvae_cvt_wT_bf16<<<(256 * FF + 255) / 256, 256, 0, stream>>>(Wd3, Wd3t, 256, FF);

    // 2) codebook norms
    vqvae_w2_kernel<<<(KK + 255) / 256, 256, 0, stream>>>(cb, w2v);

    dim3 blk(256);
    // 3) encoder MLP (layer1: fp32 A with in-LDS hw cvt; rest: bf16 A async)
    vqvae_gemm_bias_act<0, 0, 1><<<dim3(256 / TN, BB / TM), blk, 0, stream>>>(
        x, nullptr, We1t, be1, nullptr, h1b, BB, FF, 256);
    vqvae_gemm_bias_act<0, 1, 1><<<dim3(128 / TN, BB / TM), blk, 0, stream>>>(
        nullptr, h1b, We2t, be2, nullptr, h2b, BB, 256, 128);
    vqvae_gemm_bias_act<0, 1, 0><<<dim3(HH / TN, BB / TM), blk, 0, stream>>>(
        nullptr, h2b, We3t, be3, ze_out, nullptr, BB, 128, HH);

    // 4) vector-quantize (fills idx, z_q, emb, and bf16 z_q for decoder)
    vqvae_vq_kernel<<<BB, 256, 0, stream>>>(ze_out, cb, w2v, idx_out, zq_out, emb_out, zqb);

    // 5) decoder MLP
    vqvae_gemm_bias_act<0, 1, 1><<<dim3(128 / TN, BB / TM), blk, 0, stream>>>(
        nullptr, zqb, Wd1t, bd1, nullptr, g1b, BB, HH, 128);
    vqvae_gemm_bias_act<0, 1, 1><<<dim3(256 / TN, BB / TM), blk, 0, stream>>>(
        nullptr, g1b, Wd2t, bd2, nullptr, g2b, BB, 128, 256);
    vqvae_gemm_bias_act<1, 1, 0><<<dim3(FF / TN, BB / TM), blk, 0, stream>>>(
        nullptr, g2b, Wd3t, bd3, xp_out, nullptr, BB, 256, FF);
}
